// MatchingLayer_12919261626591
// MI455X (gfx1250) — matrix operand order repaired and checked by reading, not compiled, not run on hardware
//
#include <hip/hip_runtime.h>

#define HWPIX 16384
#define CDIM  256
#define NFG   4096
#define NBG   12288
#define MT    32      // rows (pixels) per workgroup
#define NC    128     // similarity columns per chunk
#define TOPK  20

typedef __attribute__((ext_vector_type(16))) __bf16 bf16x16;
typedef __attribute__((ext_vector_type(8)))  float  f32x8;

union Frag {
  uint4   q[2];   // 32 bytes
  bf16x16 v;      // 16 x bf16
};

__device__ __forceinline__ unsigned short f2bf(float x) {
  unsigned u = __float_as_uint(x);
  u += 0x7FFFu + ((u >> 16) & 1u);          // round-to-nearest-even
  return (unsigned short)(u >> 16);
}
__device__ __forceinline__ float bf2f(unsigned short h) {
  return __uint_as_float(((unsigned)h) << 16);
}

// ---------------------------------------------------------------------------
// Kernel 1: normalize s_feat [C, HW] -> row-major bf16 hi/lo [HW, C]
// ---------------------------------------------------------------------------
__global__ void prep_s_kernel(const float* __restrict__ s_feat,
                              unsigned short* __restrict__ sHi,
                              unsigned short* __restrict__ sLo) {
  __shared__ float tile[32][CDIM + 1];
  __shared__ float scale[32];
  const int t  = threadIdx.x;
  const int p0 = blockIdx.x * 32;
  const int pl = t & 31, grp = t >> 5;
  for (int c0 = 0; c0 < CDIM; c0 += 8) {            // coalesced along pixels
    int c = c0 + grp;
    tile[pl][c] = s_feat[(size_t)c * HWPIX + p0 + pl];
  }
  __syncthreads();
  if (t < 32) {
    float s = 0.f;
    for (int c = 0; c < CDIM; ++c) { float x = tile[t][c]; s += x * x; }
    scale[t] = 1.f / fmaxf(sqrtf(s), 1e-12f);
  }
  __syncthreads();
  for (int j = 0; j < 32; ++j) {                     // coalesced bf16 stores
    float v = tile[j][t] * scale[j];
    unsigned short h = f2bf(v);
    unsigned short l = f2bf(v - bf2f(h));
    size_t idx = (size_t)(p0 + j) * CDIM + t;
    sHi[idx] = h; sLo[idx] = l;
  }
}

// ---------------------------------------------------------------------------
// Kernel 2: normalize fg/bg [N, C] -> K-major bf16 hi/lo [C, N] (B operand)
// ---------------------------------------------------------------------------
__global__ void prep_coll_kernel(const float* __restrict__ fg,
                                 const float* __restrict__ bg,
                                 unsigned short* __restrict__ fgTH,
                                 unsigned short* __restrict__ fgTL,
                                 unsigned short* __restrict__ bgTH,
                                 unsigned short* __restrict__ bgTL) {
  __shared__ float tile[32][CDIM + 1];
  __shared__ float scale[32];
  const int t = threadIdx.x;
  int row0 = blockIdx.x * 32;
  const float* src; unsigned short *dh, *dl; int N, n0;
  if (row0 < NFG) { src = fg; dh = fgTH; dl = fgTL; N = NFG; n0 = row0; }
  else            { src = bg; dh = bgTH; dl = bgTL; N = NBG; n0 = row0 - NFG; }
  for (int j = 0; j < 32; ++j)                       // coalesced row loads
    tile[j][t] = src[(size_t)(n0 + j) * CDIM + t];
  __syncthreads();
  if (t < 32) {
    float s = 0.f;
    for (int c = 0; c < CDIM; ++c) { float x = tile[t][c]; s += x * x; }
    scale[t] = 1.f / fmaxf(sqrtf(s), 1e-12f);
  }
  __syncthreads();
  const int nl = t & 31, kg = t >> 5;
  for (int j = 0; j < 32; ++j) {                     // transposed stores
    int k = j * 8 + kg;
    float v = tile[nl][k] * scale[nl];
    unsigned short h = f2bf(v);
    unsigned short l = f2bf(v - bf2f(h));
    size_t idx = (size_t)k * N + n0 + nl;
    dh[idx] = h; dl[idx] = l;
  }
}

// ---------------------------------------------------------------------------
// Kernel 3: fused split-bf16 WMMA GEMM + streaming top-20 mean
// grid.x = HW/MT row tiles, grid.y = {0:fg, 1:bg}; 256 threads = 8 wave32
// ---------------------------------------------------------------------------
__global__ void __launch_bounds__(256) match_topk_kernel(
    const unsigned short* __restrict__ sHi,
    const unsigned short* __restrict__ sLo,
    const unsigned short* __restrict__ fgTH,
    const unsigned short* __restrict__ fgTL,
    const unsigned short* __restrict__ bgTH,
    const unsigned short* __restrict__ bgTL,
    float* __restrict__ out) {
  __shared__ uint4 sAH[(MT * CDIM) / 8];   // 16 KB  A-hi, bf16 row-major
  __shared__ uint4 sAL[(MT * CDIM) / 8];   // 16 KB  A-lo
  __shared__ float simBuf[MT * NC];        // 16 KB  fp32 similarity chunk

  const int t    = threadIdx.x;
  const int lane = t & 31;
  const int wave = t >> 5;
  const int which = blockIdx.y;
  const unsigned short* bHp = which ? bgTH : fgTH;
  const unsigned short* bLp = which ? bgTL : fgTL;
  const int N  = which ? NBG : NFG;
  const int p0 = blockIdx.x * MT;

  // stage this tile's 32 A rows (hi+lo) into LDS once; reused for all of N
  {
    const uint4* gh = (const uint4*)(sHi + (size_t)p0 * CDIM);
    const uint4* gl = (const uint4*)(sLo + (size_t)p0 * CDIM);
    for (int i = t; i < (MT * CDIM) / 8; i += 256) { sAH[i] = gh[i]; sAL[i] = gl[i]; }
  }
  __syncthreads();

  // per-lane running top-20 for the 4 rows this wave owns (registers)
  float topv[4][TOPK];
  float curMin[4];
  int   curMinIdx[4];
  #pragma unroll
  for (int i = 0; i < 4; ++i) {
    #pragma unroll
    for (int q = 0; q < TOPK; ++q) topv[i][q] = -1e30f;
    curMin[i] = -1e30f; curMinIdx[i] = 0;
  }

  const uint4* bh4 = (const uint4*)bHp;
  const uint4* bl4 = (const uint4*)bLp;
  const int nChunks = N / NC;

  for (int ch = 0; ch < nChunks; ++ch) {
    const int col0 = ch * NC;
    if (ch + 1 < nChunks) {                       // warm next B chunk in L2
      __builtin_prefetch(bHp + (size_t)lane * N + col0 + NC + (wave << 4), 0, 1);
      __builtin_prefetch(bLp + (size_t)lane * N + col0 + NC + (wave << 4), 0, 1);
    }
    // ---- GEMM phase: 2x8 = 16 subtiles of 16x16; 2 per wave (uniform) ----
    for (int st = wave; st < 16; st += 8) {
      const int mrow0 = (st >> 3) << 4;
      const int nt    = (st & 7) << 4;
      f32x8 acc = {};
      const int arow  = mrow0 + (lane & 15);
      const int khalf = (lane >> 4) << 3;         // A half-lane K split
      #pragma unroll
      for (int kb = 0; kb < CDIM; kb += 32) {
        Frag ah, al, bhf, blf;
        const int aidx = (arow * CDIM + kb + khalf) >> 3;  // uint4 units
        ah.q[0] = sAH[aidx]; ah.q[1] = sAH[aidx + 2];      // K +0..7, +16..23
        al.q[0] = sAL[aidx]; al.q[1] = sAL[aidx + 2];
        const size_t bcol = (size_t)col0 + nt + (lane & 15);
        #pragma unroll
        for (int s = 0; s < 16; ++s) {
          const size_t bidx = (size_t)(kb + khalf + s + (s >= 8 ? 8 : 0)) * N + bcol;
          ((unsigned short*)&bhf)[s] = bHp[bidx];
          ((unsigned short*)&blf)[s] = bLp[bidx];
        }
        // (hi+lo)*(hi+lo) ~= hi*hi + hi*lo + lo*hi  (near-fp32 dot products)
        acc = __builtin_amdgcn_wmma_f32_16x16x32_bf16(false, ah.v, false, bhf.v, (short)0, acc, false, false);
        acc = __builtin_amdgcn_wmma_f32_16x16x32_bf16(false, ah.v, false, blf.v, (short)0, acc, false, false);
        acc = __builtin_amdgcn_wmma_f32_16x16x32_bf16(false, al.v, false, bhf.v, (short)0, acc, false, false);
      }
      // D layout: lane = N (low half M0..7, high half M8..15), vgpr = M%8
      const int scol = nt + (lane & 15);
      const int srow = mrow0 + ((lane >> 4) << 3);
      #pragma unroll
      for (int e = 0; e < 8; ++e) simBuf[(srow + e) * NC + scol] = acc[e];
    }
    __syncthreads();
    // ---- selection phase: wave owns rows wave*4 .. wave*4+3 ----
    #pragma unroll
    for (int i = 0; i < 4; ++i) {
      const int r = wave * 4 + i;
      #pragma unroll
      for (int j = 0; j < NC / 32; ++j) {
        float v = simBuf[r * NC + lane + (j << 5)];
        if (v > curMin[i]) {
          const int mi = curMinIdx[i];
          #pragma unroll
          for (int q = 0; q < TOPK; ++q) if (q == mi) topv[i][q] = v;
          float m = topv[i][0]; int qi = 0;
          #pragma unroll
          for (int q = 1; q < TOPK; ++q)
            if (topv[i][q] < m) { m = topv[i][q]; qi = q; }
          curMin[i] = m; curMinIdx[i] = qi;
        }
      }
    }
    __syncthreads();
  }

  // ---- merge 32 lanes x 20 -> global top-20 mean per row ----
  #pragma unroll
  for (int i = 0; i < 4; ++i) {
    const int r = wave * 4 + i;
    float sum = 0.f;
    for (int it = 0; it < TOPK; ++it) {
      float lm = topv[i][0]; int li = 0;
      #pragma unroll
      for (int q = 1; q < TOPK; ++q)
        if (topv[i][q] > lm) { lm = topv[i][q]; li = q; }
      float wm = lm;
      #pragma unroll
      for (int off = 16; off > 0; off >>= 1)
        wm = fmaxf(wm, __shfl_xor(wm, off, 32));
      sum += wm;
      unsigned long long b = __ballot(lm == wm);
      int srcLane = __ffsll((unsigned long long)b) - 1;
      if (lane == srcLane) {                      // remove one instance
        #pragma unroll
        for (int q = 0; q < TOPK; ++q) if (q == li) topv[i][q] = -1e30f;
      }
    }
    if (lane == 0)
      out[(size_t)which * HWPIX + p0 + r] = sum * (1.0f / TOPK);
  }
}

// ---------------------------------------------------------------------------
extern "C" void kernel_launch(void* const* d_in, const int* in_sizes, int n_in,
                              void* d_out, int out_size, void* d_ws, size_t ws_size,
                              hipStream_t stream) {
  (void)in_sizes; (void)n_in; (void)out_size; (void)ws_size;
  const float* s_feat = (const float*)d_in[0];
  const float* fg     = (const float*)d_in[1];
  const float* bg     = (const float*)d_in[2];
  float* out = (float*)d_out;

  // workspace layout (32 MB total)
  unsigned char* w = (unsigned char*)d_ws;
  size_t off = 0;
  unsigned short* sHi  = (unsigned short*)(w + off); off += (size_t)HWPIX * CDIM * 2;
  unsigned short* sLo  = (unsigned short*)(w + off); off += (size_t)HWPIX * CDIM * 2;
  unsigned short* fgTH = (unsigned short*)(w + off); off += (size_t)CDIM * NFG * 2;
  unsigned short* fgTL = (unsigned short*)(w + off); off += (size_t)CDIM * NFG * 2;
  unsigned short* bgTH = (unsigned short*)(w + off); off += (size_t)CDIM * NBG * 2;
  unsigned short* bgTL = (unsigned short*)(w + off); off += (size_t)CDIM * NBG * 2;

  prep_s_kernel<<<HWPIX / 32, 256, 0, stream>>>(s_feat, sHi, sLo);
  prep_coll_kernel<<<(NFG + NBG) / 32, 256, 0, stream>>>(fg, bg, fgTH, fgTL, bgTH, bgTL);
  match_topk_kernel<<<dim3(HWPIX / MT, 2), 256, 0, stream>>>(sHi, sLo, fgTH, fgTL,
                                                             bgTH, bgTL, out);
}
